// Net_60292750901878
// MI455X (gfx1250) — matrix operand order repaired and checked by reading, not compiled, not run on hardware
//
#include <hip/hip_runtime.h>
#include <hip/hip_bf16.h>
#include <math.h>

typedef _Float16 half_t;
typedef _Float16 v16h __attribute__((ext_vector_type(16)));
typedef float    v8f  __attribute__((ext_vector_type(8)));

#define LDSS 40                      // LDS row stride in halves (80B -> conflict-free, 16B-aligned)
static constexpr float EPS_SINK = 0.01f;
static constexpr float INV_EPS  = 100.0f;
static constexpr float LMU_F    = -6.9314615f;   // log(1/1024 + 1e-8)
static constexpr float LN1024   = 6.9314718056f;
static constexpr float BN_EPS_F = 1e-5f;

// ---------------------------------------------------------------------------
// CDNA5 async memory->LDS copy (ASYNCcnt-tracked), ISA 15.18.3 op 98.
// vdst = per-lane LDS byte address, vaddr = 64-bit global address (GV mode).
// ---------------------------------------------------------------------------
__device__ __forceinline__ unsigned lds_off(const void* p) {
  return (unsigned)(uintptr_t)(const __attribute__((address_space(3))) void*)p;
}
__device__ __forceinline__ void async_ld_b128(unsigned lds_byte_off, const void* g) {
  unsigned long long ga = (unsigned long long)(uintptr_t)g;
  asm volatile("global_load_async_to_lds_b128 %0, %1, off"
               :: "v"(lds_byte_off), "v"(ga)
               : "memory");
}
__device__ __forceinline__ void wait_async0() {
  asm volatile("s_wait_asynccnt 0x0" ::: "memory");
}

// ---------------------------------------------------------------------------
// WMMA fragment helpers (ISA 7.12.2 layouts, wave32)
// A 16x32 f16: lane lo = row M; VGPR r<4 -> K = hi*8 + 2r{,+1}; r>=4 -> +16
// B 32x16 f16 stored [N][K] in LDS: lane lo = col N; VGPR r<4 -> K = hi*8 + 2r{,+1}; r>=4 -> +16
// C/D 16x16 f32: VGPR r -> row hi*8 + r, col lo
// ---------------------------------------------------------------------------
__device__ __forceinline__ v16h load_afrag(const half_t* As, int sm, int lo, int hi) {
  v16h a;
  const half_t* p = As + (sm * 16 + lo) * LDSS;
#pragma unroll
  for (int r = 0; r < 8; ++r) {
    int kb = ((r & 4) ? 16 : 0) + hi * 8 + (r & 3) * 2;
    a[2 * r]     = p[kb];
    a[2 * r + 1] = p[kb + 1];
  }
  return a;
}
__device__ __forceinline__ v16h load_bfrag(const half_t* Bs, int sn, int lo, int hi) {
  v16h b;
  const half_t* p = Bs + (sn * 16 + lo) * LDSS;
#pragma unroll
  for (int r = 0; r < 8; ++r) {
    int kb = ((r & 4) ? 16 : 0) + hi * 8 + (r & 3) * 2;
    b[2 * r]     = p[kb];
    b[2 * r + 1] = p[kb + 1];
  }
  return b;
}
__device__ __forceinline__ v8f wmma16(v16h a, v16h b, v8f c) {
  return __builtin_amdgcn_wmma_f32_16x16x32_f16(false, a, false, b, (short)0, c, false, false);
}

// ---------------------------------------------------------------------------
// Cost GEMM: C[b,i,j] = 1 - xn[b,i,:] . yn[b,j,:]   (M=N=1024, K=512)
// 128x64 block tile, 2x2 register blocking per wave, async double-buffered LDS.
// ---------------------------------------------------------------------------
__global__ __launch_bounds__(256) void k_cost_gemm(const half_t* __restrict__ xn,
                                                   const half_t* __restrict__ yn,
                                                   float* __restrict__ Cmat) {
  __shared__ half_t As[2][128 * LDSS];
  __shared__ half_t Bs[2][64 * LDSS];
  const int tid = threadIdx.x;
  const int b = blockIdx.z, m0 = blockIdx.y * 128, n0 = blockIdx.x * 64;
  const int lane = tid & 31, lo = lane & 15, hi = lane >> 4;
  const int w = tid >> 5, sr0 = (w & 3) * 2, scc = (w >> 2) * 2;
  const int arow = tid >> 1, akoff = (tid & 1) * 16;   // A: 128 rows x 32k, 16 halves/thread
  const int brow = tid >> 2, bko = (tid & 3) * 8;      // B: 64 rows x 32k, 8 halves/thread
  const half_t* Ab = xn + (size_t)b * 1024 * 512 + (size_t)(m0 + arow) * 512 + akoff;
  const half_t* Bb = yn + (size_t)b * 1024 * 512 + (size_t)(n0 + brow) * 512 + bko;

  auto issue = [&](int s, int buf) {
    const half_t* asrc = Ab + s * 32;
    unsigned ao = lds_off(&As[buf][arow * LDSS + akoff]);
    async_ld_b128(ao, asrc);
    async_ld_b128(ao + 16, asrc + 8);
    async_ld_b128(lds_off(&Bs[buf][brow * LDSS + bko]), Bb + s * 32);
  };

  v8f acc[2][2] = {};
  issue(0, 0);
  wait_async0();
  __syncthreads();
  for (int s = 0; s < 16; ++s) {
    const int cur = s & 1;
    if (s + 1 < 16) issue(s + 1, cur ^ 1);
    const half_t* Ac = As[cur];
    const half_t* Bc = Bs[cur];
    v16h a0 = load_afrag(Ac, sr0, lo, hi);
    v16h a1 = load_afrag(Ac, sr0 + 1, lo, hi);
    v16h b0 = load_bfrag(Bc, scc, lo, hi);
    v16h b1 = load_bfrag(Bc, scc + 1, lo, hi);
    acc[0][0] = wmma16(a0, b0, acc[0][0]);
    acc[0][1] = wmma16(a0, b1, acc[0][1]);
    acc[1][0] = wmma16(a1, b0, acc[1][0]);
    acc[1][1] = wmma16(a1, b1, acc[1][1]);
    wait_async0();
    __syncthreads();
  }
  float* Cb = Cmat + (size_t)b * 1024 * 1024;
#pragma unroll
  for (int da = 0; da < 2; ++da) {
    const int r0 = m0 + (sr0 + da) * 16 + hi * 8;
#pragma unroll
    for (int r = 0; r < 8; ++r) {
      Cb[(size_t)(r0 + r) * 1024 + n0 + scc * 16 + lo]       = 1.0f - acc[da][0][r];
      Cb[(size_t)(r0 + r) * 1024 + n0 + (scc + 1) * 16 + lo] = 1.0f - acc[da][1][r];
    }
  }
}

// ---------------------------------------------------------------------------
// t GEMM: t[b,n,d] = sum_m exp((u_n + v_m - C[n,m])/eps + ln1024) * sT[b,d,m]
// pi is generated on the fly in the A-tile loader. (M=1024, N=512, K=1024)
// ---------------------------------------------------------------------------
__global__ __launch_bounds__(256) void k_pi_gemm(const float* __restrict__ Cmat,
                                                 const float* __restrict__ u,
                                                 const float* __restrict__ v,
                                                 const half_t* __restrict__ sT,
                                                 float* __restrict__ t_out) {
  __shared__ half_t As[64 * LDSS];
  __shared__ half_t Bs[64 * LDSS];
  const int tid = threadIdx.x;
  const int b = blockIdx.z, m0 = blockIdx.y * 64, n0 = blockIdx.x * 64;
  const int lane = tid & 31, lo = lane & 15, hi = lane >> 4;
  const int w = tid >> 5, sm = w & 3, sc0 = (w >> 2) * 2;
  const int lrow = tid >> 2, lko = (tid & 3) * 8;
  const float* Cb = Cmat + (size_t)b * 1024 * 1024;
  const float* vb = v + (b << 10);
  const half_t* Bb = sT + (size_t)b * 512 * 1024;
  const float ui = u[(b << 10) + m0 + lrow];
  v8f acc0 = {}, acc1 = {};
  for (int k0 = 0; k0 < 1024; k0 += 32) {
    {
      const float* cp = Cb + (size_t)(m0 + lrow) * 1024 + k0 + lko;
      const float* vp = vb + k0 + lko;
      half_t tmp[8] __attribute__((aligned(16)));
#pragma unroll
      for (int e = 0; e < 8; ++e)
        tmp[e] = (half_t)__expf((ui + vp[e] - cp[e]) * INV_EPS + LN1024);
      *reinterpret_cast<uint4*>(&As[lrow * LDSS + lko]) =
          *reinterpret_cast<const uint4*>(tmp);
    }
    *reinterpret_cast<uint4*>(&Bs[lrow * LDSS + lko]) =
        *reinterpret_cast<const uint4*>(Bb + (size_t)(n0 + lrow) * 1024 + k0 + lko);
    __syncthreads();
    v16h af = load_afrag(As, sm, lo, hi);
    acc0 = wmma16(af, load_bfrag(Bs, sc0, lo, hi), acc0);
    acc1 = wmma16(af, load_bfrag(Bs, sc0 + 1, lo, hi), acc1);
    __syncthreads();
  }
  float* Ob = t_out + (size_t)b * 1024 * 512;
  const int r0 = m0 + sm * 16 + hi * 8;
#pragma unroll
  for (int r = 0; r < 8; ++r) {
    Ob[(size_t)(r0 + r) * 512 + n0 + sc0 * 16 + lo]       = acc0[r];
    Ob[(size_t)(r0 + r) * 512 + n0 + (sc0 + 1) * 16 + lo] = acc1[r];
  }
}

// ---------------------------------------------------------------------------
// Conv as 9 shifted NT-GEMMs: out[n,oc] = sum_tap sum_ic xin[shift(n),ic]*wp[tap,oc,ic]
// xin NHWC f16, wp [tap][oc][ic] f16, out NHWC fp32 (+bias, opt. ReLU)
// 128x64 block tile, async double-buffered LDS; padded rows zero-filled via DS.
// ---------------------------------------------------------------------------
__global__ __launch_bounds__(256) void k_conv_gemm(const half_t* __restrict__ xin,
                                                   const half_t* __restrict__ wp,
                                                   const float* __restrict__ bias,
                                                   float* __restrict__ out,
                                                   int IC, int OC, int relu) {
  __shared__ half_t As[2][128 * LDSS];
  __shared__ half_t Bs[2][64 * LDSS];
  const int tid = threadIdx.x;
  const int m0 = blockIdx.y * 128, n0 = blockIdx.x * 64;
  const int lane = tid & 31, lo = lane & 15, hi = lane >> 4;
  const int w = tid >> 5, sr0 = (w & 3) * 2, scc = (w >> 2) * 2;
  const int arow = tid >> 1, akoff = (tid & 1) * 16;
  const int brow = tid >> 2, bko = (tid & 3) * 8;
  const int g = m0 + arow, bb = g >> 10, hc = (g >> 5) & 31, wc = g & 31;
  const int ksteps = IC >> 5;
  const int nsteps = 9 * ksteps;

  auto issue = [&](int s, int buf) {
    const int tap = s / ksteps;
    const int k0 = (s - tap * ksteps) << 5;
    const int dy = tap / 3 - 1, dx = tap - (tap / 3) * 3 - 1;
    const int hh = hc + dy, ww = wc + dx;
    unsigned ao = lds_off(&As[buf][arow * LDSS + akoff]);
    if (hh >= 0 && hh < 32 && ww >= 0 && ww < 32) {
      const half_t* src = xin + (size_t)((bb << 10) + hh * 32 + ww) * IC + k0 + akoff;
      async_ld_b128(ao, src);
      async_ld_b128(ao + 16, src + 8);
    } else {
      const uint4 z = make_uint4(0u, 0u, 0u, 0u);
      *reinterpret_cast<uint4*>(&As[buf][arow * LDSS + akoff])     = z;
      *reinterpret_cast<uint4*>(&As[buf][arow * LDSS + akoff + 8]) = z;
    }
    const half_t* wsrc = wp + (size_t)tap * OC * IC + (size_t)(n0 + brow) * IC + k0 + bko;
    async_ld_b128(lds_off(&Bs[buf][brow * LDSS + bko]), wsrc);
  };

  v8f acc[2][2] = {};
  issue(0, 0);
  wait_async0();
  __syncthreads();
  for (int s = 0; s < nsteps; ++s) {
    const int cur = s & 1;
    if (s + 1 < nsteps) issue(s + 1, cur ^ 1);
    const half_t* Ac = As[cur];
    const half_t* Bc = Bs[cur];
    v16h a0 = load_afrag(Ac, sr0, lo, hi);
    v16h a1 = load_afrag(Ac, sr0 + 1, lo, hi);
    v16h b0 = load_bfrag(Bc, scc, lo, hi);
    v16h b1 = load_bfrag(Bc, scc + 1, lo, hi);
    acc[0][0] = wmma16(a0, b0, acc[0][0]);
    acc[0][1] = wmma16(a0, b1, acc[0][1]);
    acc[1][0] = wmma16(a1, b0, acc[1][0]);
    acc[1][1] = wmma16(a1, b1, acc[1][1]);
    wait_async0();
    __syncthreads();
  }
  const int c0 = n0 + scc * 16 + lo, c1 = c0 + 16;
  const float bv0 = bias[c0], bv1 = bias[c1];
#pragma unroll
  for (int da = 0; da < 2; ++da) {
    const int r0 = m0 + (sr0 + da) * 16 + hi * 8;
#pragma unroll
    for (int r = 0; r < 8; ++r) {
      float x0 = acc[da][0][r] + bv0, x1 = acc[da][1][r] + bv1;
      if (relu) { x0 = fmaxf(x0, 0.0f); x1 = fmaxf(x1, 0.0f); }
      out[(size_t)(r0 + r) * OC + c0] = x0;
      out[(size_t)(r0 + r) * OC + c1] = x1;
    }
  }
}

// ---------------------------------------------------------------------------
// Sinkhorn half-iterations (stable online logsumexp)
// ---------------------------------------------------------------------------
__global__ __launch_bounds__(256) void k_sink_row(const float* __restrict__ Cmat,
                                                  float* __restrict__ u,
                                                  const float* __restrict__ v) {
  __shared__ float sm_[256], ss_[256];
  const int bi = blockIdx.x, b = bi >> 10, tid = threadIdx.x;
  const float* Crow = Cmat + (size_t)bi * 1024;
  const float* vb = v + (b << 10);
  const float ui = u[bi];
  float m = -INFINITY, s = 0.0f;
  for (int j = tid; j < 1024; j += 256) {
    float x = (ui + vb[j] - Crow[j]) * INV_EPS;
    if (x <= m) s += __expf(x - m);
    else { s = s * __expf(m - x) + 1.0f; m = x; }
  }
  sm_[tid] = m; ss_[tid] = s; __syncthreads();
  for (int st = 128; st; st >>= 1) {
    if (tid < st) {
      float m1 = sm_[tid], s1 = ss_[tid], m2 = sm_[tid + st], s2 = ss_[tid + st];
      if (m2 > m1) { float t = m1; m1 = m2; m2 = t; t = s1; s1 = s2; s2 = t; }
      sm_[tid] = m1; ss_[tid] = s1 + s2 * __expf(m2 - m1);
    }
    __syncthreads();
  }
  if (!tid) u[bi] = EPS_SINK * (LMU_F - (sm_[0] + __logf(ss_[0]))) + ui;
}

__global__ __launch_bounds__(256) void k_sink_col(const float* __restrict__ Cmat,
                                                  const float* __restrict__ u,
                                                  float* __restrict__ v) {
  __shared__ float us[1024];
  __shared__ float sm_[256], ss_[256];
  const int b = blockIdx.x >> 4, jg = blockIdx.x & 15, tid = threadIdx.x;
  const int tx = tid & 63, ty = tid >> 6;
  for (int q = tid; q < 1024; q += 256) us[q] = u[(b << 10) + q];
  __syncthreads();
  const int j = jg * 64 + tx;
  const float vj = v[(b << 10) + j];
  const float* Cb = Cmat + ((size_t)b << 20) + j;
  float m = -INFINITY, s = 0.0f;
  for (int i = ty; i < 1024; i += 4) {
    float x = (us[i] + vj - Cb[(size_t)i * 1024]) * INV_EPS;
    if (x <= m) s += __expf(x - m);
    else { s = s * __expf(m - x) + 1.0f; m = x; }
  }
  sm_[tid] = m; ss_[tid] = s; __syncthreads();
  if (ty == 0) {
    float m1 = sm_[tx], s1 = ss_[tx];
#pragma unroll
    for (int q = 1; q < 4; ++q) {
      float m2 = sm_[q * 64 + tx], s2 = ss_[q * 64 + tx];
      if (m2 > m1) { float t = m1; m1 = m2; m2 = t; t = s1; s1 = s2; s2 = t; }
      s1 += s2 * __expf(m2 - m1);
    }
    v[(b << 10) + j] = EPS_SINK * (LMU_F - (m1 + __logf(s1))) + vj;
  }
}

// ---------------------------------------------------------------------------
// Small glue kernels
// ---------------------------------------------------------------------------
__global__ void k_zero(float* p, int n) {
  int i = blockIdx.x * 256 + threadIdx.x;
  if (i < n) p[i] = 0.0f;
}

// per-point inverse L2 norm over 512 strided channels (NCHW)
__global__ void k_norms(const float* __restrict__ feat, float* __restrict__ rn) {
  __shared__ float sb[256];
  const int bi = blockIdx.x, b = bi >> 10, i = bi & 1023, tid = threadIdx.x;
  const float* p = feat + (size_t)b * 512 * 1024 + i;
  float s = 0.0f;
  for (int d = tid; d < 512; d += 256) { float x = p[(size_t)d * 1024]; s += x * x; }
  sb[tid] = s; __syncthreads();
  for (int st = 128; st; st >>= 1) { if (tid < st) sb[tid] += sb[tid + st]; __syncthreads(); }
  if (!tid) rn[bi] = rsqrtf(sb[0]);
}

// NCHW fp32 -> NHWC f16 with per-point scale
__global__ void k_tnorm(const float* __restrict__ feat, const float* __restrict__ rn,
                        half_t* __restrict__ out, int total) {
  int idx = blockIdx.x * 256 + threadIdx.x;
  if (idx >= total) return;
  int d = idx & 511, i = (idx >> 9) & 1023, b = idx >> 19;
  out[idx] = (half_t)(feat[((size_t)(b * 512 + d)) * 1024 + i] * rn[(b << 10) + i]);
}

__global__ void k_f16copy(const float* __restrict__ in, half_t* __restrict__ out, int total) {
  int idx = blockIdx.x * 256 + threadIdx.x;
  if (idx < total) out[idx] = (half_t)in[idx];
}

// weights [oc][ic][3][3] fp32 -> [tap][oc][ic] f16
__global__ void k_repack(const float* __restrict__ wsrc, half_t* __restrict__ wp,
                         int OC, int IC, int total) {
  int idx = blockIdx.x * 256 + threadIdx.x;
  if (idx >= total) return;
  int tap = idx % 9, rem = idx / 9;
  int ic = rem % IC, oc = rem / IC;
  wp[((size_t)tap * OC + oc) * IC + ic] = (half_t)wsrc[idx];
}

// instance mean / rsqrt(var+eps) per (b,c), contiguous 1024 spatial
__global__ void k_cstats(const float* __restrict__ feat, float* __restrict__ cm,
                         float* __restrict__ crs) {
  __shared__ float s1[256], s2[256];
  const int bc = blockIdx.x, tid = threadIdx.x;
  const float* p = feat + (size_t)bc * 1024;
  float a = 0.0f, q = 0.0f;
  for (int e = tid; e < 1024; e += 256) { float x = p[e]; a += x; q += x * x; }
  s1[tid] = a; s2[tid] = q; __syncthreads();
  for (int st = 128; st; st >>= 1) {
    if (tid < st) { s1[tid] += s1[tid + st]; s2[tid] += s2[tid + st]; }
    __syncthreads();
  }
  if (!tid) {
    float m = s1[0] * (1.0f / 1024.0f);
    float var = s2[0] * (1.0f / 1024.0f) - m * m;
    cm[bc] = m;
    crs[bc] = rsqrtf(var + BN_EPS_F);
  }
}

// concat([t, content]) -> NHWC f16 [4][1024][1024]
__global__ void k_concat(const float* __restrict__ t_nhwc, const float* __restrict__ content,
                         half_t* __restrict__ xin, int total) {
  int idx = blockIdx.x * 256 + threadIdx.x;
  if (idx >= total) return;
  int c = idx & 1023, n = (idx >> 10) & 1023, b = idx >> 20;
  float v;
  if (c < 512) v = t_nhwc[((size_t)(b << 10) + n) * 512 + c];
  else         v = content[((size_t)(b * 512 + (c - 512))) * 1024 + n];
  xin[idx] = (half_t)v;
}

// deterministic BN batch stats: one block per channel, NHWC fp32 input
__global__ void k_bnstats(const float* __restrict__ y, float* __restrict__ bsum,
                          float* __restrict__ bsq, int C) {
  __shared__ float s1[256], s2[256];
  const int c = blockIdx.x, tid = threadIdx.x;
  float a = 0.0f, q = 0.0f;
  for (int r = tid; r < 4096; r += 256) { float x = y[(size_t)r * C + c]; a += x; q += x * x; }
  s1[tid] = a; s2[tid] = q; __syncthreads();
  for (int st = 128; st; st >>= 1) {
    if (tid < st) { s1[tid] += s1[tid + st]; s2[tid] += s2[tid + st]; }
    __syncthreads();
  }
  if (!tid) { bsum[c] = s1[0]; bsq[c] = s2[0]; }
}

__global__ void k_bnapply(const float* __restrict__ y, const float* __restrict__ bsum,
                          const float* __restrict__ bsq, const float* __restrict__ g,
                          const float* __restrict__ be, half_t* __restrict__ out,
                          int C, int total) {
  int idx = blockIdx.x * 256 + threadIdx.x;
  if (idx >= total) return;
  int c = idx % C;
  float m = bsum[c] * (1.0f / 4096.0f);
  float var = bsq[c] * (1.0f / 4096.0f) - m * m;
  float sc = g[c] * rsqrtf(var + BN_EPS_F);
  out[idx] = (half_t)((y[idx] - m) * sc + be[c]);
}

// t = alpha * (content - mean)/std + t, stored NHWC f16 for decoder
__global__ void k_combine(const float* __restrict__ alpha, const float* __restrict__ t_nhwc,
                          const float* __restrict__ content, const float* __restrict__ cm,
                          const float* __restrict__ crs, half_t* __restrict__ tt, int total) {
  int idx = blockIdx.x * 256 + threadIdx.x;
  if (idx >= total) return;
  int d = idx & 511, ng = idx >> 9;
  int b = ng >> 10, nl = ng & 1023;
  int bc = b * 512 + d;
  float val = alpha[idx] * (content[(size_t)bc * 1024 + nl] - cm[bc]) * crs[bc] + t_nhwc[idx];
  tt[idx] = (half_t)val;
}

// decoder conv 512 -> 3, direct (0.11 GFLOP), output NCHW fp32
__global__ void k_dec(const half_t* __restrict__ tt, const float* __restrict__ dw,
                      const float* __restrict__ db, float* __restrict__ out) {
  int idx = blockIdx.x * 256 + threadIdx.x;
  if (idx >= 4 * 3 * 1024) return;
  int wq = idx & 31, hq = (idx >> 5) & 31;
  int oc = (idx >> 10) % 3, b = idx / 3072;
  float acc = db[oc];
  for (int ky = 0; ky < 3; ++ky) {
    int hh = hq + ky - 1;
    if (hh < 0 || hh > 31) continue;
    for (int kx = 0; kx < 3; ++kx) {
      int ww = wq + kx - 1;
      if (ww < 0 || ww > 31) continue;
      const half_t* xp = tt + (size_t)((b << 10) + hh * 32 + ww) * 512;
      const float* wp = dw + (size_t)oc * 512 * 9 + ky * 3 + kx;
      float s = 0.0f;
      for (int ic = 0; ic < 512; ++ic) s += (float)xp[ic] * wp[(size_t)ic * 9];
      acc += s;
    }
  }
  out[idx] = acc;
}

// ---------------------------------------------------------------------------
extern "C" void kernel_launch(void* const* d_in, const int* in_sizes, int n_in,
                              void* d_out, int out_size, void* d_ws, size_t ws_size,
                              hipStream_t stream) {
  const float* content = (const float*)d_in[0];
  const float* style   = (const float*)d_in[1];
  const float* w1  = (const float*)d_in[2];  const float* b1  = (const float*)d_in[3];
  const float* g1  = (const float*)d_in[4];  const float* be1 = (const float*)d_in[5];
  const float* w2  = (const float*)d_in[6];  const float* b2  = (const float*)d_in[7];
  const float* g2  = (const float*)d_in[8];  const float* be2 = (const float*)d_in[9];
  const float* w3  = (const float*)d_in[10]; const float* b3  = (const float*)d_in[11];
  const float* dcw = (const float*)d_in[12]; const float* dcb = (const float*)d_in[13];
  float* out = (float*)d_out;

  // workspace layout (256B aligned)
  char* base = (char*)d_ws;
  size_t off = 0;
  auto alloc = [&](size_t bytes) -> char* {
    off = (off + 255) & ~(size_t)255;
    char* p = base + off;
    off += bytes;
    return p;
  };
  half_t* xn_h   = (half_t*)alloc(4ull * 1024 * 512 * 2);     // 4 MB
  half_t* yn_h   = (half_t*)alloc(4ull * 1024 * 512 * 2);     // 4 MB
  half_t* sT_h   = (half_t*)alloc(4ull * 512 * 1024 * 2);     // 4 MB
  float*  Cmat   = (float*) alloc(4ull * 1024 * 1024 * 4);    // 16 MB
  float*  uv     = (float*) alloc(2ull * 4096 * 4);           // u+v
  float*  u      = uv;
  float*  v      = uv + 4096;
  float*  rn_c   = (float*) alloc(4096 * 4);
  float*  rn_s   = (float*) alloc(4096 * 4);
  float*  t_nhwc = (float*) alloc(4ull * 1024 * 512 * 4);     // 8 MB
  half_t* xin    = (half_t*)alloc(4ull * 1024 * 1024 * 2);    // 8 MB
  half_t* wp1    = (half_t*)alloc(9ull * 256 * 1024 * 2);     // 4.5 MB
  half_t* wp2    = (half_t*)alloc(9ull * 256 * 256 * 2);      // 1.125 MB
  half_t* wp3    = (half_t*)alloc(9ull * 512 * 256 * 2);      // 2.25 MB
  float*  cm     = (float*) alloc(2048 * 4);
  float*  crs    = (float*) alloc(2048 * 4);
  float*  bnsum  = (float*) alloc(256 * 4);
  float*  bnsq   = (float*) alloc(256 * 4);
  half_t* tt     = (half_t*)alloc(4096ull * 512 * 2);         // 4 MB
  // aliases: regions dead after the Sinkhorn phase are reused by the convs
  float*  y1   = (float*)xn_h;                 // 4096x256 fp32  (4 MB)
  half_t* y1h  = (half_t*)yn_h;                // 4096x256 f16   (2 MB)
  float*  y2   = (float*)sT_h;                 // 4096x256 fp32  (4 MB)
  half_t* y2h  = (half_t*)Cmat;                // 4096x256 f16   (2 MB)
  float*  alpha = (float*)((char*)Cmat + (4ull << 20)); // 4096x512 fp32 (8 MB)

  // --- weight repack (independent of data path) ---
  k_repack<<<(256 * 1024 * 9 + 255) / 256, 256, 0, stream>>>(w1, wp1, 256, 1024, 256 * 1024 * 9);
  k_repack<<<(256 * 256 * 9 + 255) / 256, 256, 0, stream>>>(w2, wp2, 256, 256, 256 * 256 * 9);
  k_repack<<<(512 * 256 * 9 + 255) / 256, 256, 0, stream>>>(w3, wp3, 512, 256, 512 * 256 * 9);

  // --- normalize + transpose features to NHWC f16 ---
  k_norms<<<4096, 256, 0, stream>>>(content, rn_c);
  k_norms<<<4096, 256, 0, stream>>>(style, rn_s);
  k_tnorm<<<2097152 / 256, 256, 0, stream>>>(content, rn_c, xn_h, 2097152);
  k_tnorm<<<2097152 / 256, 256, 0, stream>>>(style, rn_s, yn_h, 2097152);
  k_f16copy<<<2097152 / 256, 256, 0, stream>>>(style, sT_h, 2097152);

  // --- cosine cost matrix via WMMA (async double-buffered) ---
  k_cost_gemm<<<dim3(16, 8, 4), 256, 0, stream>>>(xn_h, yn_h, Cmat);

  // --- Sinkhorn iterations ---
  k_zero<<<8192 / 256, 256, 0, stream>>>(uv, 8192);
  for (int it = 0; it < 20; ++it) {
    k_sink_row<<<4096, 256, 0, stream>>>(Cmat, u, v);
    k_sink_col<<<64, 256, 0, stream>>>(Cmat, u, v);
  }

  // --- t = (pi * 1024) @ sf via WMMA, pi fused in A-loader ---
  k_pi_gemm<<<dim3(8, 16, 4), 256, 0, stream>>>(Cmat, u, v, sT_h, t_nhwc);

  // --- instance stats of content ---
  k_cstats<<<2048, 256, 0, stream>>>(content, cm, crs);

  // --- alpha predictor: conv/BN stack, all WMMA ---
  k_concat<<<4194304 / 256, 256, 0, stream>>>(t_nhwc, content, xin, 4194304);
  k_conv_gemm<<<dim3(4, 32), 256, 0, stream>>>(xin, wp1, b1, y1, 1024, 256, 1);
  k_bnstats<<<256, 256, 0, stream>>>(y1, bnsum, bnsq, 256);
  k_bnapply<<<1048576 / 256, 256, 0, stream>>>(y1, bnsum, bnsq, g1, be1, y1h, 256, 1048576);
  k_conv_gemm<<<dim3(4, 32), 256, 0, stream>>>(y1h, wp2, b2, y2, 256, 256, 1);
  k_bnstats<<<256, 256, 0, stream>>>(y2, bnsum, bnsq, 256);
  k_bnapply<<<1048576 / 256, 256, 0, stream>>>(y2, bnsum, bnsq, g2, be2, y2h, 256, 1048576);
  k_conv_gemm<<<dim3(8, 32), 256, 0, stream>>>(y2h, wp3, b3, alpha, 256, 512, 0);

  // --- combine and decode ---
  k_combine<<<2097152 / 256, 256, 0, stream>>>(alpha, t_nhwc, content, cm, crs, tt, 2097152);
  k_dec<<<48, 256, 0, stream>>>(tt, dcw, dcb, out);
}
